// BipartiteGCNEncoder_57827439674232
// MI455X (gfx1250) — matrix operand order repaired and checked by reading, not compiled, not run on hardware
//
#include <hip/hip_runtime.h>

#define H 128
#define NVAR 100000
#define NCON 50000
#define NEDGE 1000000
#define BN_EPS 1e-5f

typedef __attribute__((ext_vector_type(16))) __bf16 bf16x16;
typedef __attribute__((ext_vector_type(8)))  __bf16 bf16x8;
typedef __attribute__((ext_vector_type(8)))  float  f32x8;

// ---------------------------------------------------------------------------
// WMMA fragment loaders (contiguous k order for A and B, wave32)
// A (16x32 bf16, from LDS row-major tile):
//   lane<16 : M=lane,    K = k0+0..15
//   lane>=16: M=lane-16, K = k0+16..31
// B (32x16 bf16, from transposed weights WT[N][Kpad]):
//   lane<16 : N=n0+lane,    K=k0+0..15   (contiguous in WT row)
//   lane>=16: N=n0+lane-16, K=k0+16..31
// C/D (16x16 f32): vgpr j, lane l -> M = j + 8*(l>>4), N = (l&15)
// ---------------------------------------------------------------------------
__device__ __forceinline__ bf16x16 load_a_frag(const __bf16* tile, int stride,
                                               int k0, int lane) {
  const int m  = lane & 15;
  const int hi = lane >> 4;
  const __bf16* p = tile + m * stride + k0 + 16 * hi;
  bf16x8 lo = *(const bf16x8*)(p);
  bf16x8 hv = *(const bf16x8*)(p + 8);
  bf16x16 r;
#pragma unroll
  for (int i = 0; i < 8; ++i) { r[i] = lo[i]; r[i + 8] = hv[i]; }
  return r;
}

__device__ __forceinline__ bf16x16 load_b_frag(const __bf16* wt, int kpad,
                                               int n0, int k0, int lane) {
  const int n  = n0 + (lane & 15);
  const int kb = k0 + 16 * (lane >> 4);
  return *(const bf16x16*)(wt + (long)n * kpad + kb);
}

// Convert 8 consecutive f32 (32B-aligned) to a packed bf16x8.
__device__ __forceinline__ bf16x8 cvt8(const float* __restrict__ p) {
  const float4 f0 = *(const float4*)(p);
  const float4 f1 = *(const float4*)(p + 4);
  bf16x8 r;
  r[0] = (__bf16)f0.x; r[1] = (__bf16)f0.y; r[2] = (__bf16)f0.z; r[3] = (__bf16)f0.w;
  r[4] = (__bf16)f1.x; r[5] = (__bf16)f1.y; r[6] = (__bf16)f1.z; r[7] = (__bf16)f1.w;
  return r;
}

// ---------------------------------------------------------------------------
// 16xN GEMM tile: acc[8] (16x128) += A(16xKTOT from LDS) @ WT(KTOTx128).
// B-fragments are batched per k-step (8 distinct regs -> one wait per 8 WMMAs)
// and double-buffered across k-steps so L2 latency overlaps the WMMA pipe.
// ---------------------------------------------------------------------------
template <int KTOT>
__device__ __forceinline__ void gemm_tile(const __bf16* atile, int astride,
                                          const __bf16* __restrict__ wt, int kpad,
                                          int lane, f32x8 acc[8]) {
  bf16x16 bcur[8], bnxt[8];
#pragma unroll
  for (int nt = 0; nt < 8; ++nt) bcur[nt] = load_b_frag(wt, kpad, nt * 16, 0, lane);
#pragma unroll
  for (int k0 = 0; k0 < KTOT; k0 += 32) {
    bf16x16 a = load_a_frag(atile, astride, k0, lane);
    if (k0 + 32 < KTOT) {
#pragma unroll
      for (int nt = 0; nt < 8; ++nt)
        bnxt[nt] = load_b_frag(wt, kpad, nt * 16, k0 + 32, lane);
    }
#pragma unroll
    for (int nt = 0; nt < 8; ++nt)
      acc[nt] = __builtin_amdgcn_wmma_f32_16x16x32_bf16(
          false, a, false, bcur[nt], (short)0, acc[nt], false, false);
#pragma unroll
    for (int nt = 0; nt < 8; ++nt) bcur[nt] = bnxt[nt];
  }
}

// ---------------------------------------------------------------------------
// Weight convert: W[K][128] f32 -> WT[128][Kpad] bf16 (transposed, zero pad)
// ---------------------------------------------------------------------------
__global__ void wconv_kernel(const float* __restrict__ W, __bf16* __restrict__ WT,
                             int K, int Kpad) {
  int idx = blockIdx.x * blockDim.x + threadIdx.x;
  int n = idx / Kpad, k = idx % Kpad;
  if (n >= H) return;
  WT[(long)n * Kpad + k] = (k < K) ? (__bf16)W[(long)k * H + n] : (__bf16)0.0f;
}

// ---------------------------------------------------------------------------
// Embedding: out = relu(x @ W + b), x[N][D], W[D][128]. One row per block.
// ---------------------------------------------------------------------------
__global__ void embed_kernel(const float* __restrict__ x, const float* __restrict__ W,
                             const float* __restrict__ b, float* __restrict__ out,
                             int N, int D) {
  long row = blockIdx.x;
  int n = threadIdx.x;
  if (row >= N) return;
  float acc = b[n];
  for (int k = 0; k < D; ++k) acc += x[row * D + k] * W[k * H + n];
  out[row * H + n] = fmaxf(acc, 0.0f);
}

// ---------------------------------------------------------------------------
// Fused edge message kernel:
//   per wave: 16 edges. A = [src_h | dst_h | edge_attr] (16x257, pad->288) bf16
//   hidden = relu(A @ g1 + b1); msg = hidden @ g2 + b2; atomicAdd agg[dst]
// ---------------------------------------------------------------------------
__global__ void __launch_bounds__(128)
edge_message_kernel(const float* __restrict__ srcH, const float* __restrict__ dstH,
                    const int* __restrict__ srcIdx, const int* __restrict__ dstIdx,
                    const float* __restrict__ edgeAttr,
                    const __bf16* __restrict__ g1T, const float* __restrict__ gb1,
                    const __bf16* __restrict__ g2T, const float* __restrict__ gb2,
                    float* __restrict__ agg, int E) {
  __shared__ __align__(32) __bf16 sA[4][16][288];
  __shared__ __align__(32) __bf16 sHid[4][16][128];
  __shared__ int sDst[4][16];

  const int lane = threadIdx.x & 31;
  const int wave = threadIdx.x >> 5;
  const long base = ((long)blockIdx.x * 4 + wave) * 16;
  if (base >= E) return;  // wave-uniform guard -> EXEC all-1s below

  const int hi = lane >> 4;
  const int nl = lane & 15;

  // ---- gather A tile: cols [0,128)=src row, [128,256)=dst row, 256=edge ----
  const int c0 = lane * 8;
  for (int r = 0; r < 16; ++r) {
    const long e = base + r;
    const int s = srcIdx[e];
    const int d = dstIdx[e];
    if (lane == 0) sDst[wave][r] = d;
    const float* p = (c0 < H) ? (srcH + (long)s * H + c0)
                              : (dstH + (long)d * H + (c0 - H));
    *(bf16x8*)(&sA[wave][r][c0]) = cvt8(p);
    if (lane == 0) {
      __bf16* arow = &sA[wave][r][0];
      arow[256] = (__bf16)edgeAttr[e];
#pragma unroll
      for (int i = 257; i < 288; ++i) arow[i] = (__bf16)0.0f;
    }
  }

  // ---- GEMM 1: hidden = relu(A @ g1 + b1) ----
  f32x8 acc[8];
#pragma unroll
  for (int nt = 0; nt < 8; ++nt)
#pragma unroll
    for (int j = 0; j < 8; ++j) acc[nt][j] = 0.0f;

  gemm_tile<288>(&sA[wave][0][0], 288, g1T, 288, lane, acc);

#pragma unroll
  for (int nt = 0; nt < 8; ++nt) {
    const int n = nt * 16 + nl;
    const float bias = gb1[n];
#pragma unroll
    for (int j = 0; j < 8; ++j) {
      float v = acc[nt][j] + bias;
      sHid[wave][j + 8 * hi][n] = (__bf16)(v > 0.0f ? v : 0.0f);
    }
  }

  // ---- GEMM 2: msg = hidden @ g2 + b2 ----
#pragma unroll
  for (int nt = 0; nt < 8; ++nt)
#pragma unroll
    for (int j = 0; j < 8; ++j) acc[nt][j] = 0.0f;

  gemm_tile<128>(&sHid[wave][0][0], 128, g2T, 128, lane, acc);

  // ---- scatter-add to per-destination aggregates ----
#pragma unroll
  for (int nt = 0; nt < 8; ++nt) {
    const int n = nt * 16 + nl;
    const float bias = gb2[n];
#pragma unroll
    for (int j = 0; j < 8; ++j) {
      const int m = j + 8 * hi;
      const int d = sDst[wave][m];
      atomicAdd(agg + (long)d * H + n, acc[nt][j] + bias);
    }
  }
}

// ---------------------------------------------------------------------------
// BatchNorm stats: per-column sum / sumsq over agg[N][128]
// ---------------------------------------------------------------------------
__global__ void bn_stats_kernel(const float* __restrict__ agg,
                                float* __restrict__ sums, float* __restrict__ sumsq,
                                int N) {
  const int n = threadIdx.x;  // blockDim = 128
  float s = 0.0f, s2 = 0.0f;
  for (long r = blockIdx.x; r < N; r += gridDim.x) {
    float v = agg[r * H + n];
    s += v; s2 += v * v;
  }
  atomicAdd(sums + n, s);
  atomicAdd(sumsq + n, s2);
}

__global__ void bn_finalize_kernel(const float* __restrict__ sums,
                                   const float* __restrict__ sumsq,
                                   const float* __restrict__ gamma,
                                   const float* __restrict__ beta,
                                   float* __restrict__ scale,
                                   float* __restrict__ shift, float invN) {
  const int n = threadIdx.x;  // 128 threads, 1 block
  const float mean = sums[n] * invN;
  const float var = sumsq[n] * invN - mean * mean;
  const float sc = gamma[n] * rsqrtf(var + BN_EPS);
  scale[n] = sc;
  shift[n] = beta[n] - mean * sc;
}

// ---------------------------------------------------------------------------
// Node update: out = relu([dst | BN(agg)] @ f1 + fb1) @ f2 + fb2
// ---------------------------------------------------------------------------
__global__ void __launch_bounds__(128)
node_update_kernel(const float* __restrict__ dstH, const float* __restrict__ agg,
                   const float* __restrict__ scale, const float* __restrict__ shift,
                   const __bf16* __restrict__ f1T, const float* __restrict__ fb1,
                   const __bf16* __restrict__ f2T, const float* __restrict__ fb2,
                   float* __restrict__ out, int N) {
  __shared__ __align__(32) __bf16 sA[4][16][256];
  __shared__ __align__(32) __bf16 sHid[4][16][128];

  const int lane = threadIdx.x & 31;
  const int wave = threadIdx.x >> 5;
  const long base = ((long)blockIdx.x * 4 + wave) * 16;
  if (base >= N) return;  // wave-uniform

  const int hi = lane >> 4;
  const int nl = lane & 15;

  // ---- build A tile: [dst_h | agg*scale + shift] ----
  const int c0 = lane * 8;
  for (int r = 0; r < 16; ++r) {
    const long row = base + r;
    if (c0 < H) {
      *(bf16x8*)(&sA[wave][r][c0]) = cvt8(dstH + row * H + c0);
    } else {
      const int cc = c0 - H;
      const float* p = agg + row * H + cc;
      bf16x8 t;
#pragma unroll
      for (int i = 0; i < 8; ++i)
        t[i] = (__bf16)(p[i] * scale[cc + i] + shift[cc + i]);
      *(bf16x8*)(&sA[wave][r][c0]) = t;
    }
  }

  f32x8 acc[8];
#pragma unroll
  for (int nt = 0; nt < 8; ++nt)
#pragma unroll
    for (int j = 0; j < 8; ++j) acc[nt][j] = 0.0f;

  gemm_tile<256>(&sA[wave][0][0], 256, f1T, 256, lane, acc);

#pragma unroll
  for (int nt = 0; nt < 8; ++nt) {
    const int n = nt * 16 + nl;
    const float bias = fb1[n];
#pragma unroll
    for (int j = 0; j < 8; ++j) {
      float v = acc[nt][j] + bias;
      sHid[wave][j + 8 * hi][n] = (__bf16)(v > 0.0f ? v : 0.0f);
    }
  }

#pragma unroll
  for (int nt = 0; nt < 8; ++nt)
#pragma unroll
    for (int j = 0; j < 8; ++j) acc[nt][j] = 0.0f;

  gemm_tile<128>(&sHid[wave][0][0], 128, f2T, 128, lane, acc);

#pragma unroll
  for (int nt = 0; nt < 8; ++nt) {
    const int n = nt * 16 + nl;
    const float bias = fb2[n];
#pragma unroll
    for (int j = 0; j < 8; ++j) {
      const int m = j + 8 * hi;
      out[(base + m) * H + n] = acc[nt][j] + bias;
    }
  }
}

// ---------------------------------------------------------------------------
extern "C" void kernel_launch(void* const* d_in, const int* in_sizes, int n_in,
                              void* d_out, int out_size, void* d_ws, size_t ws_size,
                              hipStream_t stream) {
  // setup_inputs() flatten order (insertion order):
  const float* x_var    = (const float*)d_in[0];
  const float* x_con    = (const float*)d_in[1];
  const int*   con_idx  = (const int*)d_in[2];
  const int*   var_idx  = (const int*)d_in[3];
  const float* edge_attr= (const float*)d_in[4];
  const float* var_W = (const float*)d_in[5];
  const float* var_b = (const float*)d_in[6];
  const float* con_W = (const float*)d_in[7];
  const float* con_b = (const float*)d_in[8];
  // v2c params [9..18], c2v params [19..28]: gW1,gb1,gW2,gb2,fW1,fb1,fW2,fb2,gamma,beta
  const float* v2c[10]; const float* c2v[10];
  for (int i = 0; i < 10; ++i) { v2c[i] = (const float*)d_in[9 + i]; c2v[i] = (const float*)d_in[19 + i]; }

  float* out = (float*)d_out;
  float* out_v = out;                      // [NVAR][128]
  float* out_c = out + (long)NVAR * H;     // [NCON][128]

  // bump allocator on workspace
  size_t off = 0;
  auto alloc = [&](size_t bytes) -> char* {
    off = (off + 255) & ~(size_t)255;
    char* p = (char*)d_ws + off;
    off += bytes;
    return p;
  };
  float*  v    = (float*)alloc((size_t)NVAR * H * 4);
  float*  c    = (float*)alloc((size_t)NCON * H * 4);
  float*  agg  = (float*)alloc((size_t)NVAR * H * 4);  // sized for the larger pass
  __bf16* g1T_v2c = (__bf16*)alloc((size_t)H * 288 * 2);
  __bf16* g2T_v2c = (__bf16*)alloc((size_t)H * 128 * 2);
  __bf16* f1T_v2c = (__bf16*)alloc((size_t)H * 256 * 2);
  __bf16* f2T_v2c = (__bf16*)alloc((size_t)H * 128 * 2);
  __bf16* g1T_c2v = (__bf16*)alloc((size_t)H * 288 * 2);
  __bf16* g2T_c2v = (__bf16*)alloc((size_t)H * 128 * 2);
  __bf16* f1T_c2v = (__bf16*)alloc((size_t)H * 256 * 2);
  __bf16* f2T_c2v = (__bf16*)alloc((size_t)H * 128 * 2);
  float* stats = (float*)alloc(4 * H * 4);
  float* sums = stats, *sumsq = stats + H, *scale = stats + 2 * H, *shift = stats + 3 * H;

  // ---- convert weights to bf16, transposed + padded ----
  wconv_kernel<<<(H * 288 + 255) / 256, 256, 0, stream>>>(v2c[0], g1T_v2c, 257, 288);
  wconv_kernel<<<(H * 128 + 255) / 256, 256, 0, stream>>>(v2c[2], g2T_v2c, 128, 128);
  wconv_kernel<<<(H * 256 + 255) / 256, 256, 0, stream>>>(v2c[4], f1T_v2c, 256, 256);
  wconv_kernel<<<(H * 128 + 255) / 256, 256, 0, stream>>>(v2c[6], f2T_v2c, 128, 128);
  wconv_kernel<<<(H * 288 + 255) / 256, 256, 0, stream>>>(c2v[0], g1T_c2v, 257, 288);
  wconv_kernel<<<(H * 128 + 255) / 256, 256, 0, stream>>>(c2v[2], g2T_c2v, 128, 128);
  wconv_kernel<<<(H * 256 + 255) / 256, 256, 0, stream>>>(c2v[4], f1T_c2v, 256, 256);
  wconv_kernel<<<(H * 128 + 255) / 256, 256, 0, stream>>>(c2v[6], f2T_c2v, 128, 128);

  // ---- embeddings ----
  embed_kernel<<<NVAR, H, 0, stream>>>(x_var, var_W, var_b, v, NVAR, 19);
  embed_kernel<<<NCON, H, 0, stream>>>(x_con, con_W, con_b, c, NCON, 5);

  const int edge_blocks = (NEDGE / 16 + 3) / 4;  // 4 tiles (waves) per block

  // ================= pass 1: V -> C =================
  hipMemsetAsync(agg, 0, (size_t)NCON * H * 4, stream);
  hipMemsetAsync(stats, 0, 2 * H * 4, stream);
  edge_message_kernel<<<edge_blocks, 128, 0, stream>>>(
      v, c, var_idx, con_idx, edge_attr,
      g1T_v2c, v2c[1], g2T_v2c, v2c[3], agg, NEDGE);
  bn_stats_kernel<<<1024, H, 0, stream>>>(agg, sums, sumsq, NCON);
  bn_finalize_kernel<<<1, H, 0, stream>>>(sums, sumsq, v2c[8], v2c[9], scale, shift,
                                          1.0f / (float)NCON);
  node_update_kernel<<<(NCON / 16 + 3) / 4, 128, 0, stream>>>(
      c, agg, scale, shift, f1T_v2c, v2c[5], f2T_v2c, v2c[7], out_c, NCON);

  // ================= pass 2: C -> V =================
  hipMemsetAsync(agg, 0, (size_t)NVAR * H * 4, stream);
  hipMemsetAsync(stats, 0, 2 * H * 4, stream);
  edge_message_kernel<<<edge_blocks, 128, 0, stream>>>(
      out_c, v, con_idx, var_idx, edge_attr,
      g1T_c2v, c2v[1], g2T_c2v, c2v[3], agg, NEDGE);
  bn_stats_kernel<<<1024, H, 0, stream>>>(agg, sums, sumsq, NVAR);
  bn_finalize_kernel<<<1, H, 0, stream>>>(sums, sumsq, c2v[8], c2v[9], scale, shift,
                                          1.0f / (float)NVAR);
  node_update_kernel<<<(NVAR / 16 + 3) / 4, 128, 0, stream>>>(
      v, agg, scale, shift, f1T_c2v, c2v[5], f2T_c2v, c2v[7], out_v, NVAR);
}
